// NodeProcessor_14731737825431
// MI455X (gfx1250) — matrix operand order repaired and checked by reading, not compiled, not run on hardware
//
#include <hip/hip_runtime.h>

typedef __attribute__((ext_vector_type(16))) __bf16 v16bf;
typedef __attribute__((ext_vector_type(8)))  __bf16 v8bf;
typedef __attribute__((ext_vector_type(8)))  float  v8f;

#define D_NODE   128
#define IN_DIM   256
#define HIDDEN   256
#define OUT_DIM  128
#define TM        32      // node rows per block
#define ROWSTR   264      // LDS row stride (elements), keeps 16B alignment, skews banks

// Split fp32 into bf16 hi (truncation) + bf16 lo (residual, truncated).
static __device__ __forceinline__ void split_bf16(float f, unsigned short& h, unsigned short& l) {
    unsigned u = __float_as_uint(f) & 0xFFFF0000u;
    h = (unsigned short)(u >> 16);
    float r = f - __uint_as_float(u);
    l = (unsigned short)(__float_as_uint(r) >> 16);
}

// Build a 16-element bf16 A/B fragment from two 16-byte chunks.
static __device__ __forceinline__ v16bf frag16(const unsigned short* p0, const unsigned short* p1) {
    v8bf a = *(const v8bf*)p0;
    v8bf b = *(const v8bf*)p1;
    return __builtin_shufflevector(a, b, 0,1,2,3,4,5,6,7,8,9,10,11,12,13,14,15);
}

static __device__ __forceinline__ v8f wmma_bf16(v16bf a, v16bf b, v8f c) {
    return __builtin_amdgcn_wmma_f32_16x16x32_bf16(false, a, false, b, (short)0, c, false, false);
}

// ---------------- kernel 0: zero the aggregation buffer ----------------
__global__ void gnn_zero_agg(float* agg, int n) {
    int i = blockIdx.x * 256 + threadIdx.x;
    if (i < n) agg[i] = 0.0f;
}

// ---------------- kernel 1: transpose + hi/lo split weights ----------------
// W1 [IN_DIM][HIDDEN] row-major -> w1t_{hi,lo}[n][k] (n=HIDDEN rows, k=IN_DIM contiguous)
// W2 [HIDDEN][OUT_DIM]          -> w2t_{hi,lo}[n][k] (n=OUT_DIM rows, k=HIDDEN contiguous)
__global__ void gnn_prep_weights(const float* __restrict__ W1, const float* __restrict__ W2,
                                 unsigned short* w1t_hi, unsigned short* w1t_lo,
                                 unsigned short* w2t_hi, unsigned short* w2t_lo) {
    int idx = blockIdx.x * 256 + threadIdx.x;
    if (idx < IN_DIM * HIDDEN) {
        int k = idx >> 8, n = idx & 255;             // W1[k][n]
        unsigned short h, l; split_bf16(W1[idx], h, l);
        w1t_hi[n * IN_DIM + k] = h;
        w1t_lo[n * IN_DIM + k] = l;
    } else {
        int j = idx - IN_DIM * HIDDEN;
        if (j < HIDDEN * OUT_DIM) {
            int k = j >> 7, n = j & 127;             // W2[k][n]
            unsigned short h, l; split_bf16(W2[j], h, l);
            w2t_hi[n * HIDDEN + k] = h;
            w2t_lo[n * HIDDEN + k] = l;
        }
    }
}

// ---------------- kernel 2: scatter-sum edge features ----------------
// one float4 (4 features) per thread, 32 threads per edge
__global__ void gnn_scatter(const float4* __restrict__ edge_attr,
                            const long long* __restrict__ dst,
                            float* agg, int n_edges) {
    long long t = (long long)blockIdx.x * 256 + threadIdx.x;
    int e = (int)(t >> 5);
    if (e >= n_edges) return;
    int c = (int)t & 31;
    float4 v = edge_attr[(size_t)e * 32 + c];
    long long d = dst[e];
    float* p = agg + (size_t)d * D_NODE + c * 4;
    atomicAdd(p + 0, v.x);
    atomicAdd(p + 1, v.y);
    atomicAdd(p + 2, v.z);
    atomicAdd(p + 3, v.w);
}

// ---------------- kernel 3: fused concat -> MLP -> LayerNorm -> residual ----------------
__global__ __launch_bounds__(256)
void gnn_mlp(const float* __restrict__ x, const float* __restrict__ agg,
             const unsigned short* __restrict__ w1t_hi, const unsigned short* __restrict__ w1t_lo,
             const unsigned short* __restrict__ w2t_hi, const unsigned short* __restrict__ w2t_lo,
             const float* __restrict__ b1, const float* __restrict__ b2,
             const float* __restrict__ ln_g, const float* __restrict__ ln_b,
             float* __restrict__ out, int n_nodes) {
    // s_hi/s_lo hold concat[x,agg] rows during GEMM1, then reused for relu(h1).
    __shared__ __attribute__((aligned(16))) unsigned short s_hi[TM][ROWSTR];
    __shared__ __attribute__((aligned(16))) unsigned short s_lo[TM][ROWSTR];
    __shared__ __attribute__((aligned(16))) float s_h2[TM][OUT_DIM + 4];

    const int tid     = threadIdx.x;
    const int wave    = tid >> 5;
    const int lane    = tid & 31;
    const int rt      = wave >> 2;         // row-tile (0..1): rows rt*16..rt*16+15
    const int quad    = wave & 3;          // column group
    const int r       = lane & 15;
    const int halfsel = lane >> 4;         // 0: lanes 0-15, 1: lanes 16-31
    const int c0      = halfsel * 16;      // A-fragment K sub-chunk
    const int kbh     = halfsel * 16;      // B-fragment K half
    const long long row0 = (long long)blockIdx.x * TM;

    // ---- stage concat[x, agg] rows into LDS as bf16 hi/lo ----
    for (int idx = tid; idx < TM * IN_DIM; idx += 256) {
        int rloc = idx >> 8, col = idx & 255;
        long long node = row0 + rloc;
        float v = 0.0f;
        if (node < n_nodes)
            v = (col < D_NODE) ? x[node * D_NODE + col]
                               : agg[node * D_NODE + (col - D_NODE)];
        unsigned short h, l; split_bf16(v, h, l);
        s_hi[rloc][col] = h;
        s_lo[rloc][col] = l;
    }
    __syncthreads();

    // ---- GEMM1: h1[32][256] = concat @ W1, bf16x3 split accumulation ----
    v8f acc[4] = {};
    for (int ks = 0; ks < 8; ++ks) {
        int kb = ks * 32;
        const unsigned short* ph = &s_hi[rt * 16 + r][kb + c0];
        const unsigned short* pl = &s_lo[rt * 16 + r][kb + c0];
        v16bf Ahi = frag16(ph, ph + 8);
        v16bf Alo = frag16(pl, pl + 8);
#pragma unroll
        for (int ct = 0; ct < 4; ++ct) {
            int n = quad * 64 + ct * 16 + r;
            v16bf Bhi = *(const v16bf*)(w1t_hi + n * IN_DIM + kb + kbh);
            v16bf Blo = *(const v16bf*)(w1t_lo + n * IN_DIM + kb + kbh);
            acc[ct] = wmma_bf16(Alo, Bhi, acc[ct]);
            acc[ct] = wmma_bf16(Ahi, Blo, acc[ct]);
            acc[ct] = wmma_bf16(Ahi, Bhi, acc[ct]);
        }
    }
    __syncthreads();   // everyone done reading staged inputs

    // ---- bias + ReLU, re-split h1 into LDS (reusing s_hi/s_lo) ----
#pragma unroll
    for (int ct = 0; ct < 4; ++ct) {
        int col = quad * 64 + ct * 16 + r;
        float bias = b1[col];
#pragma unroll
        for (int i = 0; i < 8; ++i) {
            int row = rt * 16 + i + halfsel * 8;
            float v = acc[ct][i] + bias;
            v = v > 0.0f ? v : 0.0f;
            unsigned short h, l; split_bf16(v, h, l);
            s_hi[row][col] = h;
            s_lo[row][col] = l;
        }
    }
    __syncthreads();

    // ---- GEMM2: h2[32][128] = h1 @ W2, bf16x3 ----
    v8f acc2[2] = {};
    for (int ks = 0; ks < 8; ++ks) {
        int kb = ks * 32;
        const unsigned short* ph = &s_hi[rt * 16 + r][kb + c0];
        const unsigned short* pl = &s_lo[rt * 16 + r][kb + c0];
        v16bf Ahi = frag16(ph, ph + 8);
        v16bf Alo = frag16(pl, pl + 8);
#pragma unroll
        for (int u = 0; u < 2; ++u) {
            int n = quad * 16 + u * 64 + r;
            v16bf Bhi = *(const v16bf*)(w2t_hi + n * HIDDEN + kb + kbh);
            v16bf Blo = *(const v16bf*)(w2t_lo + n * HIDDEN + kb + kbh);
            acc2[u] = wmma_bf16(Alo, Bhi, acc2[u]);
            acc2[u] = wmma_bf16(Ahi, Blo, acc2[u]);
            acc2[u] = wmma_bf16(Ahi, Bhi, acc2[u]);
        }
    }

    // ---- h2 + b2 into LDS (fp32) ----
#pragma unroll
    for (int u = 0; u < 2; ++u) {
        int col = quad * 16 + u * 64 + r;
        float bias = b2[col];
#pragma unroll
        for (int i = 0; i < 8; ++i) {
            int row = rt * 16 + i + halfsel * 8;
            s_h2[row][col] = acc2[u][i] + bias;
        }
    }
    __syncthreads();

    // ---- LayerNorm(OUT_DIM=128) + residual; one wave handles 4 rows ----
    for (int rloc = wave * 4; rloc < wave * 4 + 4; ++rloc) {
        float vals[4];
        float s = 0.0f, q = 0.0f;
#pragma unroll
        for (int i = 0; i < 4; ++i) {
            float v = s_h2[rloc][lane + 32 * i];
            vals[i] = v;
            s += v;
            q += v * v;
        }
        for (int off = 16; off > 0; off >>= 1) {
            s += __shfl_xor(s, off, 32);
            q += __shfl_xor(q, off, 32);
        }
        float mu  = s * (1.0f / OUT_DIM);
        float var = q * (1.0f / OUT_DIM) - mu * mu;
        float rs  = rsqrtf(var + 1e-5f);
        long long node = row0 + rloc;
        if (node < n_nodes) {
#pragma unroll
            for (int i = 0; i < 4; ++i) {
                int c = lane + 32 * i;
                float o = x[node * D_NODE + c] + (vals[i] - mu) * rs * ln_g[c] + ln_b[c];
                out[node * D_NODE + c] = o;
            }
        }
    }
}

extern "C" void kernel_launch(void* const* d_in, const int* in_sizes, int n_in,
                              void* d_out, int out_size, void* d_ws, size_t ws_size,
                              hipStream_t stream) {
    const float*     x         = (const float*)d_in[0];
    const long long* edge_idx  = (const long long*)d_in[1];   // int64 [2][E]
    const float*     edge_attr = (const float*)d_in[2];
    const float*     W1        = (const float*)d_in[3];
    const float*     b1        = (const float*)d_in[4];
    const float*     W2        = (const float*)d_in[5];
    const float*     b2        = (const float*)d_in[6];
    const float*     ln_g      = (const float*)d_in[7];
    const float*     ln_b      = (const float*)d_in[8];
    float*           out       = (float*)d_out;

    const int n_nodes = in_sizes[0] / D_NODE;
    const int n_edges = in_sizes[2] / D_NODE;

    // workspace layout
    char*  ws    = (char*)d_ws;
    float* agg   = (float*)ws;
    size_t off   = (((size_t)n_nodes * D_NODE * sizeof(float)) + 255) & ~(size_t)255;
    unsigned short* w1t_hi = (unsigned short*)(ws + off);  off += (size_t)IN_DIM * HIDDEN * 2;
    unsigned short* w1t_lo = (unsigned short*)(ws + off);  off += (size_t)IN_DIM * HIDDEN * 2;
    unsigned short* w2t_hi = (unsigned short*)(ws + off);  off += (size_t)HIDDEN * OUT_DIM * 2;
    unsigned short* w2t_lo = (unsigned short*)(ws + off);

    const long long* dst = edge_idx + n_edges;  // row 1 = destination nodes

    int agg_elems = n_nodes * D_NODE;
    gnn_zero_agg<<<(agg_elems + 255) / 256, 256, 0, stream>>>(agg, agg_elems);

    int prep_elems = IN_DIM * HIDDEN + HIDDEN * OUT_DIM;
    gnn_prep_weights<<<(prep_elems + 255) / 256, 256, 0, stream>>>(
        W1, W2, w1t_hi, w1t_lo, w2t_hi, w2t_lo);

    long long scat_threads = (long long)n_edges * 32;
    gnn_scatter<<<(unsigned)((scat_threads + 255) / 256), 256, 0, stream>>>(
        (const float4*)edge_attr, dst, agg, n_edges);

    int n_blocks = (n_nodes + TM - 1) / TM;
    gnn_mlp<<<n_blocks, 256, 0, stream>>>(
        x, agg, w1t_hi, w1t_lo, w2t_hi, w2t_lo, b1, b2, ln_g, ln_b, out, n_nodes);
}
